// MultivariateGaussian_89807766159361
// MI455X (gfx1250) — matrix operand order repaired and checked by reading, not compiled, not run on hardware
//
#include <hip/hip_runtime.h>
#include <math.h>

#define NROWS 8192
#define DDIM  512

// ---- CDNA5 WMMA types ------------------------------------------------------
typedef __attribute__((ext_vector_type(16))) __bf16       v16bf;
typedef __attribute__((ext_vector_type(8)))  float        v8f;
typedef __attribute__((ext_vector_type(4)))  unsigned int u32x4;
typedef __attribute__((ext_vector_type(4)))  __bf16       bf16x4;

constexpr int BK   = 32;  // K per WMMA / per LDS stage (bf16 elems)
constexpr int QROW = 5;   // u32x4 chunks per LDS row: 4 data (64B) + 1 skew (16B)
                          // 80B row stride -> 16B aligned, bank-conflict free

struct QPair { u32x4 lo, hi; };
static __device__ __forceinline__ v16bf mkfrag(u32x4 a, u32x4 b) {
    QPair p{a, b};
    return __builtin_bit_cast(v16bf, p);
}

struct TilesQ {
    u32x4 ahi[128 * QROW];
    u32x4 alo[128 * QROW];
    u32x4 bhi[128 * QROW];
    u32x4 blo[128 * QROW];
};                                   // 40960 B
union SmemT {
    TilesQ t;
    float  cstage[64 * 129];         // 33024 B transpose staging (half tile)
};

// ---- Kernel 1: split fp32 x into bf16 hi + bf16 lo planes ------------------
__global__ __launch_bounds__(256)
void split_bf16_kernel(const float* __restrict__ x,
                       unsigned short* __restrict__ hi_,
                       unsigned short* __restrict__ lo_) {
    __bf16* hi = (__bf16*)hi_;
    __bf16* lo = (__bf16*)lo_;
    size_t i = ((size_t)blockIdx.x * 256u + threadIdx.x) * 4u;
    float4 f = *(const float4*)(x + i);
    __bf16 h0 = (__bf16)f.x, h1 = (__bf16)f.y, h2 = (__bf16)f.z, h3 = (__bf16)f.w;
    __bf16 l0 = (__bf16)(f.x - (float)h0);
    __bf16 l1 = (__bf16)(f.y - (float)h1);
    __bf16 l2 = (__bf16)(f.z - (float)h2);
    __bf16 l3 = (__bf16)(f.w - (float)h3);
    bf16x4 hv; hv[0] = h0; hv[1] = h1; hv[2] = h2; hv[3] = h3;
    bf16x4 lv; lv[0] = l0; lv[1] = l1; lv[2] = l2; lv[3] = l3;
    *(bf16x4*)(hi + i) = hv;
    *(bf16x4*)(lo + i) = lv;
}

// ---- Kernel 2: cov = s * (x @ x^T), symmetric (upper blocks + mirror) ------
// Block = 128x128 tile of C, 256 threads = 8 waves (wave32), wave grid 2(M)x4(N),
// each wave owns 4x2 WMMA tiles. Split-precision: hi*hi + lo*hi + hi*lo.
__global__ __launch_bounds__(256)
void xxt_wmma_kernel(const unsigned short* __restrict__ hi_,
                     const unsigned short* __restrict__ lo_,
                     const float* __restrict__ rho,
                     float* __restrict__ cov) {
    const int bx = blockIdx.x, by = blockIdx.y;
    if (bx < by) return;  // symmetric: compute only upper-triangular block pairs

    __shared__ SmemT sm;
    const __bf16* __restrict__ xhi = (const __bf16*)hi_;
    const __bf16* __restrict__ xlo = (const __bf16*)lo_;

    const int tid  = threadIdx.x;
    const int lane = tid & 31, wave = tid >> 5;
    const int wm = wave >> 2, wn = wave & 3;     // 2 x 4 wave grid
    const int l16 = lane & 15, lh = lane >> 4;   // half-wave id
    const float s = rho[0];

    const size_t arow0 = (size_t)by * 128;       // C row-tile = rows of x (A)
    const size_t brow0 = (size_t)bx * 128;       // C col-tile = rows of x (B)

    // Loop-invariant LDS fragment base pointers (chunk indices become imm offsets)
    const u32x4* aBaseH = &sm.t.ahi[(wm * 64 + l16) * QROW + lh];
    const u32x4* aBaseL = &sm.t.alo[(wm * 64 + l16) * QROW + lh];
    const u32x4* bBaseH = &sm.t.bhi[(wn * 32 + l16) * QROW + lh];
    const u32x4* bBaseL = &sm.t.blo[(wn * 32 + l16) * QROW + lh];

    // Global-load coordinates for the staging phase (2 chunks per thread/plane)
    const int grow0 = tid >> 2;              // rows t/4 and t/4+64
    const int gch   = tid & 3;               // 16B chunk within 64B k-slab

    v8f acc[4][2];
    #pragma unroll
    for (int i = 0; i < 4; ++i)
        #pragma unroll
        for (int j = 0; j < 2; ++j)
            #pragma unroll
            for (int e = 0; e < 8; ++e) acc[i][j][e] = 0.0f;

    // Prefetch K-tile 0 into registers
    u32x4 pre[2][4];
    #pragma unroll
    for (int h = 0; h < 2; ++h) {
        const int row = grow0 + h * 64;
        const size_t ga = (arow0 + row) * DDIM + gch * 8;
        const size_t gb = (brow0 + row) * DDIM + gch * 8;
        pre[h][0] = *(const u32x4*)(xhi + ga);
        pre[h][1] = *(const u32x4*)(xlo + ga);
        pre[h][2] = *(const u32x4*)(xhi + gb);
        pre[h][3] = *(const u32x4*)(xlo + gb);
    }

    for (int kb = 0; kb < DDIM; kb += BK) {
        // Stage prefetched tile into LDS (ds_store_b128 x8 per thread)
        #pragma unroll
        for (int h = 0; h < 2; ++h) {
            const int so = (grow0 + h * 64) * QROW + gch;
            sm.t.ahi[so] = pre[h][0];
            sm.t.alo[so] = pre[h][1];
            sm.t.bhi[so] = pre[h][2];
            sm.t.blo[so] = pre[h][3];
        }
        __syncthreads();

        // Prefetch next K-tile while WMMAs run
        if (kb + BK < DDIM) {
            #pragma unroll
            for (int h = 0; h < 2; ++h) {
                const int row = grow0 + h * 64;
                const size_t ga = (arow0 + row) * DDIM + (kb + BK) + gch * 8;
                const size_t gb = (brow0 + row) * DDIM + (kb + BK) + gch * 8;
                pre[h][0] = *(const u32x4*)(xhi + ga);
                pre[h][1] = *(const u32x4*)(xlo + ga);
                pre[h][2] = *(const u32x4*)(xhi + gb);
                pre[h][3] = *(const u32x4*)(xlo + gb);
            }
        }

        // B fragments (32x16 K-major): row n=l%16; lane<16 -> k {0..7,16..23};
        // lane>=16 -> {8..15,24..31} -> chunks {lh, 2+lh}
        v16bf bh[2], bl[2];
        #pragma unroll
        for (int nt = 0; nt < 2; ++nt) {
            const int rb = nt * 16 * QROW;
            bh[nt] = mkfrag(bBaseH[rb], bBaseH[rb + 2]);
            bl[nt] = mkfrag(bBaseL[rb], bBaseL[rb + 2]);
        }
        // A fragments (16x32): lane<16 -> k {0..7,16..23}; lane>=16 -> {8..15,24..31}
        // -> chunks {lh, 2+lh}
        #pragma unroll
        for (int mt = 0; mt < 4; ++mt) {
            const int ra = mt * 16 * QROW;
            const v16bf ah = mkfrag(aBaseH[ra], aBaseH[ra + 2]);
            const v16bf al = mkfrag(aBaseL[ra], aBaseL[ra + 2]);
            #pragma unroll
            for (int nt = 0; nt < 2; ++nt) {
                acc[mt][nt] = __builtin_amdgcn_wmma_f32_16x16x32_bf16(
                    false, ah, false, bh[nt], (short)0, acc[mt][nt], false, false);
                acc[mt][nt] = __builtin_amdgcn_wmma_f32_16x16x32_bf16(
                    false, al, false, bh[nt], (short)0, acc[mt][nt], false, false);
                acc[mt][nt] = __builtin_amdgcn_wmma_f32_16x16x32_bf16(
                    false, ah, false, bl[nt], (short)0, acc[mt][nt], false, false);
            }
        }
        __syncthreads();
    }

    // Direct store of the (by,bx) block. D layout: vgpr r, lanes 0-15 -> M=r,
    // lanes 16-31 -> M=8+r, N=lane%16 (64B contiguous per half-wave).
    #pragma unroll
    for (int mt = 0; mt < 4; ++mt)
        #pragma unroll
        for (int nt = 0; nt < 2; ++nt)
            #pragma unroll
            for (int r = 0; r < 8; ++r) {
                const int ml = wm * 64 + mt * 16 + lh * 8 + r;
                const int nl = wn * 32 + nt * 16 + l16;
                __builtin_nontemporal_store(acc[mt][nt][r] * s,
                    &cov[(arow0 + ml) * (size_t)NROWS + (brow0 + nl)]);
            }

    // Mirror to the (bx,by) block via LDS transpose (coalesced stores).
    if (bx != by) {
        for (int half = 0; half < 2; ++half) {      // 64 rows of C-tile at a time
            __syncthreads();
            if (wm == half) {
                #pragma unroll
                for (int mt = 0; mt < 4; ++mt)
                    #pragma unroll
                    for (int nt = 0; nt < 2; ++nt)
                        #pragma unroll
                        for (int r = 0; r < 8; ++r) {
                            const int mloc = mt * 16 + lh * 8 + r;        // 0..63
                            const int nl   = wn * 32 + nt * 16 + l16;     // 0..127
                            sm.cstage[mloc * 129 + nl] = acc[mt][nt][r] * s;
                        }
            }
            __syncthreads();
            #pragma unroll
            for (int i = 0; i < 32; ++i) {
                const int idx  = tid + (i << 8);
                const int mloc = idx & 63;
                const int nl   = idx >> 6;
                __builtin_nontemporal_store(sm.cstage[mloc * 129 + nl],
                    &cov[(brow0 + nl) * (size_t)NROWS + (arow0 + half * 64 + mloc)]);
            }
        }
    }
}

// ---- Kernel 3: mu head + softplus diagonal (one wave per row) --------------
__global__ __launch_bounds__(256)
void mu_diag_kernel(const float* __restrict__ x,
                    const float* __restrict__ muk,
                    const float* __restrict__ vark,
                    const float* __restrict__ mub,
                    const float* __restrict__ varb,
                    float* __restrict__ out_mu,
                    float* __restrict__ cov) {
    const int wave = threadIdx.x >> 5;
    const int lane = threadIdx.x & 31;
    const int row  = blockIdx.x * 8 + wave;
    const float* xr = x + (size_t)row * DDIM;
    float sm = 0.f, sv = 0.f;
    #pragma unroll 4
    for (int k = lane; k < DDIM; k += 32) {
        const float xv = xr[k];
        sm += xv * muk[k];
        sv += xv * vark[k];
    }
    #pragma unroll
    for (int off = 16; off > 0; off >>= 1) {
        sm += __shfl_xor(sm, off, 32);
        sv += __shfl_xor(sv, off, 32);
    }
    if (lane == 0) {
        out_mu[row] = sm + mub[0];
        const float v  = sv + varb[0];
        const float sp = (v > 20.0f) ? v : log1pf(expf(v));
        cov[(size_t)row * NROWS + row] = sp + 1e-8f;
    }
}

// ---- Launch ----------------------------------------------------------------
extern "C" void kernel_launch(void* const* d_in, const int* in_sizes, int n_in,
                              void* d_out, int out_size, void* d_ws, size_t ws_size,
                              hipStream_t stream) {
    const float* x    = (const float*)d_in[0];   // [8192, 512]
    const float* muk  = (const float*)d_in[1];   // [512, 1]
    const float* rho  = (const float*)d_in[2];   // [1]
    const float* vark = (const float*)d_in[3];   // [512, 1]
    const float* mub  = (const float*)d_in[4];   // [1]
    const float* varb = (const float*)d_in[5];   // [1]

    float* out_mu = (float*)d_out;               // first N floats
    float* cov    = out_mu + NROWS;              // then N*N floats

    unsigned short* xhi = (unsigned short*)d_ws;                    // 8 MB bf16 hi
    unsigned short* xlo = xhi + (size_t)NROWS * DDIM;               // 8 MB bf16 lo

    split_bf16_kernel<<<dim3((NROWS * DDIM) / (256 * 4)), 256, 0, stream>>>(x, xhi, xlo);
    xxt_wmma_kernel<<<dim3(NROWS / 128, NROWS / 128), 256, 0, stream>>>(xhi, xlo, rho, cov);
    mu_diag_kernel<<<dim3(NROWS / 8), 256, 0, stream>>>(x, muk, vark, mub, varb, out_mu, cov);
}
